// RNNModel_8899172237971
// MI455X (gfx1250) — compile-verified, matrix operand order checked by reading, not run on hardware
//
#include <hip/hip_runtime.h>

typedef __attribute__((ext_vector_type(16))) _Float16 v16h;
typedef __attribute__((ext_vector_type(8)))  _Float16 v8h;
typedef __attribute__((ext_vector_type(8)))  float    v8f;
typedef __attribute__((ext_vector_type(4)))  int      v4i;

typedef __attribute__((address_space(1))) v4i g_v4i;   // global (prints as __device__)
typedef __attribute__((address_space(3))) v4i l_v4i;   // LDS

#define RNN_B 256
#define RNN_T 512
#define RNN_I 196
#define RNN_H 64
#define RNN_C 10
#define KC    7            // 7 chunks of K=32 -> Kpad = 224
#define ROWS_A 64          // M rows per workgroup in projection kernel (4 waves)
#define WFRAG (4 * KC * 512)   // packed W_ih fragment elements (14336 halves)

#if __has_builtin(__builtin_amdgcn_global_load_async_to_lds_b128) && \
    __has_builtin(__builtin_amdgcn_s_wait_asynccnt)
#define USE_ASYNC_LDS 1
#else
#define USE_ASYNC_LDS 0
#endif

// K index within a 32-wide chunk for 16-bit WMMA A/B fragments:
// k = (h&7) + ((h&8)<<1) + ((lane>>4)<<3), h = half index 0..15
__device__ __forceinline__ int frag_k(int lane, int h) {
    return (h & 7) + ((h & 8) << 1) + ((lane >> 4) << 3);
}

// ---------------------------------------------------------------------------
// Pre-pack kernel (1 block, one-time): W_ih -> WMMA B-fragment order (f16,
// K zero-padded to 224) and fused bias (b_ih + b_hh). All 2048 projection
// WGs then consume this L2-resident buffer with straight b128 copies instead
// of re-deriving the fragment shuffle (kills the div/mod VALU overhead in
// the bandwidth-bound kernel).
// ---------------------------------------------------------------------------
__global__ __launch_bounds__(256) void rnn_wpack_kernel(
    const float* __restrict__ W_ih, const float* __restrict__ b_ih,
    const float* __restrict__ b_hh, _Float16* __restrict__ WBg,
    float* __restrict__ biasg)
{
    const int tid = threadIdx.x;
    for (int s = tid; s < WFRAG; s += 256) {
        int h   = s & 15;
        int l   = (s >> 4) & 31;
        int knt = s >> 9;              // nt*7 + kc
        int kc  = knt % KC;
        int nt  = knt / KC;
        int n   = nt * 16 + (l & 15);
        int k   = kc * 32 + frag_k(l, h);
        _Float16 v = (_Float16)0.f;
        if (k < RNN_I) v = (_Float16)W_ih[n * RNN_I + k];
        WBg[s] = v;
    }
    if (tid < RNN_H) biasg[tid] = b_ih[tid] + b_hh[tid];
}

// ---------------------------------------------------------------------------
// Kernel A: xp[m, n] = sum_k x[m,k] * W_ih[n,k] + bias[n]
// m = b*T + t, M = 131072, K = 196 (pad 224), N = 64. Memory-bound: streams
// x once at HBM rate; WMMA supplies the 32 FLOP/byte the roofline demands.
// ---------------------------------------------------------------------------
__global__ __launch_bounds__(128) void rnn_xproj_kernel(
    const float* __restrict__ x, const _Float16* __restrict__ WBg,
    const float* __restrict__ biasg, float* __restrict__ xp)
{
    // Fragment-native LDS: 4 waves * 7 kc * 32 lanes * 16 halves each
    __shared__ __align__(16) _Float16 XA[WFRAG];   // 28672 B (x tile)
    __shared__ __align__(16) _Float16 WBs[WFRAG];  // 28672 B (packed W)

    const int tid  = threadIdx.x;
    const int lane = tid & 31;
    const int wave = tid >> 5;
    const int wgM  = blockIdx.x * ROWS_A;

    // Bring pre-packed W fragments into LDS.
#if USE_ASYNC_LDS
    // gfx1250 async DMA path: no VGPR round-trip, tracked by ASYNCcnt.
    for (int i = tid; i < WFRAG / 8; i += 128) {
        __builtin_amdgcn_global_load_async_to_lds_b128(
            (g_v4i*)(unsigned long long)(WBg + i * 8),
            (l_v4i*)(unsigned int)(unsigned long long)(WBs + i * 8),
            0, 0);
    }
#else
    for (int i = tid; i < WFRAG / 8; i += 128)
        *(v8h*)&WBs[i * 8] = *(const v8h*)&WBg[i * 8];
#endif

    // Zero XA (covers the K-padding slots)
    for (int s = tid; s < WFRAG / 2; s += 128)
        ((unsigned int*)XA)[s] = 0u;
    __syncthreads();

    // Scatter x tile (coalesced global reads, fp32 -> f16 into fragment layout)
    const float* xr = x + (size_t)wgM * RNN_I;
    for (int e = tid; e < ROWS_A * RNN_I; e += 128) {
        int row = e / RNN_I;
        int k   = e - row * RNN_I;
        _Float16 v = (_Float16)xr[e];
        int w  = row >> 4;
        int l  = (row & 15) | (((k >> 3) & 1) << 4);
        int h  = (k & 7) | (((k >> 4) & 1) << 3);
        int kc = k >> 5;
        XA[((w * KC + kc) * 32 + l) * 16 + h] = v;
    }
#if USE_ASYNC_LDS
    __builtin_amdgcn_s_wait_asynccnt(0);
#endif
    __syncthreads();

    // Each wave: its 16 rows x all 64 cols, K accumulated over 7 chunks
    v16h a[KC];
#pragma unroll
    for (int kc = 0; kc < KC; ++kc)
        a[kc] = *(const v16h*)&XA[((wave * KC + kc) * 32 + lane) * 16];

#pragma unroll
    for (int nt = 0; nt < 4; ++nt) {
        v8f c = {0.f, 0.f, 0.f, 0.f, 0.f, 0.f, 0.f, 0.f};
#pragma unroll
        for (int kc = 0; kc < KC; ++kc) {
            v16h b = *(const v16h*)&WBs[((nt * KC + kc) * 32 + lane) * 16];
            c = __builtin_amdgcn_wmma_f32_16x16x32_f16(
                    false, a[kc], false, b, (short)0, c, false, false);
        }
        int   n    = nt * 16 + (lane & 15);
        float bias = biasg[n];
        int   m0   = wgM + wave * 16 + ((lane >> 4) << 3);
#pragma unroll
        for (int r = 0; r < 8; ++r)
            xp[(size_t)(m0 + r) * RNN_H + n] = c[r] + bias;
    }
}

// ---------------------------------------------------------------------------
// Kernel B: sequential scan h_t = tanh(xp_t + h_{t-1} W_hh^T), then FC.
// 16 independent WGs (16 batch rows each); 4 waves = 4 N-tiles; W_hh
// fragments pinned in registers for all 512 steps; h double-buffered in LDS
// (1 barrier/step); xp used directly as the WMMA C accumulator.
// ---------------------------------------------------------------------------
__global__ __launch_bounds__(128) void rnn_scan_kernel(
    const float* __restrict__ xp, const float* __restrict__ W_hh,
    const float* __restrict__ fc_W, const float* __restrict__ fc_b,
    float* __restrict__ out)
{
    __shared__ __align__(16) _Float16 hbuf[2][16 * 64];
    __shared__ float hfin[16 * 64];

    const int tid  = threadIdx.x;
    const int lane = tid & 31;
    const int nt   = tid >> 5;         // wave id == N-tile id
    const int bm0  = blockIdx.x * 16;  // first batch row of this WG
    const int rhi  = (lane >> 4) << 3; // row/K-run selector from lane-half
    const int ncol = nt * 16 + (lane & 15);

    // h_0 = 0
    for (int s = tid; s < 16 * 64; s += 128) hbuf[0][s] = (_Float16)0.f;

    // Persistent W_hh B-fragments (K = 64 -> 2 chunks of 32)
    v16h wf[2];
    {
        int n = ncol;
#pragma unroll
        for (int kc = 0; kc < 2; ++kc)
#pragma unroll
            for (int j = 0; j < 16; ++j)
                wf[kc][j] = (_Float16)W_hh[n * RNN_H + kc * 32 + frag_k(lane, j)];
    }
    __syncthreads();

    int buf = 0;
    for (int t = 0; t < RNN_T; ++t) {
        // A fragments from row-major f16 h (two contiguous 16-B runs each)
        const _Float16* hb = hbuf[buf];
        const int row = lane & 15;
        v8h lo0 = *(const v8h*)&hb[row * 64 + rhi];
        v8h hi0 = *(const v8h*)&hb[row * 64 + rhi + 16];
        v8h lo1 = *(const v8h*)&hb[row * 64 + 32 + rhi];
        v8h hi1 = *(const v8h*)&hb[row * 64 + 32 + rhi + 16];
        v16h a0, a1;
#pragma unroll
        for (int j = 0; j < 8; ++j) {
            a0[j] = lo0[j]; a0[j + 8] = hi0[j];
            a1[j] = lo1[j]; a1[j + 8] = hi1[j];
        }

        // C accumulator = xp tile for timestep t (L2-resident)
        const float* xpt = xp + ((size_t)(bm0 + rhi) * RNN_T + t) * RNN_H + ncol;
        v8f c;
#pragma unroll
        for (int r = 0; r < 8; ++r)
            c[r] = xpt[(size_t)r * RNN_T * RNN_H];
        if (t + 1 < RNN_T)
            __builtin_prefetch(xpt + RNN_H, 0, 1);   // next step -> global_prefetch_b8

        v8f d = __builtin_amdgcn_wmma_f32_16x16x32_f16(
                    false, a0, false, wf[0], (short)0, c, false, false);
        d = __builtin_amdgcn_wmma_f32_16x16x32_f16(
                    false, a1, false, wf[1], (short)0, d, false, false);

        // tanh(x) = 1 - 2/(exp(2x)+1)  (saturates cleanly, no NaN at +/-inf)
        _Float16* hnew = hbuf[buf ^ 1];
#pragma unroll
        for (int r = 0; r < 8; ++r) {
            float e  = __expf(2.f * d[r]);
            float th = 1.f - 2.f / (e + 1.f);
            hnew[(rhi + r) * 64 + ncol] = (_Float16)th;
            if (t == RNN_T - 1) hfin[(rhi + r) * 64 + ncol] = th;
        }
        __syncthreads();
        buf ^= 1;
    }

    // Final FC on h_T (f32 copy): out[b, cc] = h_T[b,:] . fc_W[cc,:] + fc_b[cc]
    for (int s = tid; s < 16 * RNN_C; s += 128) {
        int bl = s / RNN_C;
        int cc = s - bl * RNN_C;
        float acc = fc_b[cc];
#pragma unroll
        for (int k = 0; k < RNN_H; ++k)
            acc += hfin[bl * 64 + k] * fc_W[cc * RNN_H + k];
        out[(size_t)(bm0 + bl) * RNN_C + cc] = acc;
    }
}

// ---------------------------------------------------------------------------
extern "C" void kernel_launch(void* const* d_in, const int* in_sizes, int n_in,
                              void* d_out, int out_size, void* d_ws, size_t ws_size,
                              hipStream_t stream) {
    const float* x    = (const float*)d_in[0];
    const float* W_ih = (const float*)d_in[1];
    const float* W_hh = (const float*)d_in[2];
    const float* b_ih = (const float*)d_in[3];
    const float* b_hh = (const float*)d_in[4];
    const float* fc_W = (const float*)d_in[5];
    const float* fc_b = (const float*)d_in[6];
    float* out = (float*)d_out;

    // Workspace layout:
    //   xp    : [B*T, H] f32  = 33,554,432 B
    //   WBg   : packed W_ih fragments, 14336 f16 = 28,672 B
    //   biasg : fused bias, 64 f32 = 256 B
    float*    xp    = (float*)d_ws;
    _Float16* WBg   = (_Float16*)((char*)d_ws + (size_t)RNN_B * RNN_T * RNN_H * sizeof(float));
    float*    biasg = (float*)((char*)WBg + (size_t)WFRAG * sizeof(_Float16));

    const int M = RNN_B * RNN_T;                  // 131072
    rnn_wpack_kernel<<<1, 256, 0, stream>>>(W_ih, b_ih, b_hh, WBg, biasg);
    rnn_xproj_kernel<<<M / ROWS_A, 128, 0, stream>>>(x, WBg, biasg, xp);
    rnn_scan_kernel<<<RNN_B / 16, 128, 0, stream>>>(xp, W_hh, fc_W, fc_b, out);
}
